// node_mlp_out_5162550689856
// MI455X (gfx1250) — compile-verified, matrix operand order checked by reading, not run on hardware
//
#include <hip/hip_runtime.h>
#include <hip/hip_bf16.h>

// ---------------------------------------------------------------------------
// Problem constants (from the reference)
// ---------------------------------------------------------------------------
enum : int {
    N_NODES = 40000,
    N_EDGES = 640000,
    NODE_F  = 128,
    EDGE_F  = 128,
    HIDDEN  = 1024,
    OUT_F   = 256,
    D_IN    = NODE_F + EDGE_F  // 256
};

typedef __bf16  bf16x16 __attribute__((ext_vector_type(16)));
typedef __bf16  bf16x8  __attribute__((ext_vector_type(8)));
typedef __bf16  bf16x4  __attribute__((ext_vector_type(4)));
typedef float   floatx8 __attribute__((ext_vector_type(8)));

// ---------------------------------------------------------------------------
// 1) zero fp32 accumulator (float4 granularity)
// ---------------------------------------------------------------------------
__global__ void zero_f4_kernel(float4* __restrict__ p, long n4) {
    long i = (long)blockIdx.x * blockDim.x + threadIdx.x;
    if (i < n4) p[i] = make_float4(0.f, 0.f, 0.f, 0.f);
}

// ---------------------------------------------------------------------------
// 2) scatter-add edge_attr over source nodes: agg[row[e]] += edge_attr[e]
//    one thread handles one (edge, 4-float chunk) -> 4 global_atomic_add_f32
// ---------------------------------------------------------------------------
__global__ void scatter_add_kernel(const float* __restrict__ edge_attr,
                                   const int*   __restrict__ rows,
                                   float*       __restrict__ agg,
                                   int nE) {
    int idx = blockIdx.x * blockDim.x + threadIdx.x;
    int e = idx >> 5;                 // 128 feats / 4 = 32 chunks per edge
    if (e >= nE) return;
    int q = idx & 31;
    int r = rows[e];
    float4 v = ((const float4*)edge_attr)[(long)e * 32 + q];
    float* dst = agg + (long)r * EDGE_F + q * 4;
    atomicAdd(dst + 0, v.x);
    atomicAdd(dst + 1, v.y);
    atomicAdd(dst + 2, v.z);
    atomicAdd(dst + 3, v.w);
}

// ---------------------------------------------------------------------------
// 3) pack h0 = concat([x, agg], axis=1) as bf16, row-major 40000 x 256
// ---------------------------------------------------------------------------
__global__ void pack_h0_kernel(const float* __restrict__ x,
                               const float* __restrict__ agg,
                               __bf16*      __restrict__ h0,
                               int nNodes) {
    long i = (long)blockIdx.x * blockDim.x + threadIdx.x;
    long total = (long)nNodes * (D_IN / 4);
    if (i >= total) return;
    int node = (int)(i >> 6);         // 256/4 = 64 chunks per node
    int t    = (int)(i & 63);
    int c    = t * 4;
    const float* src = (c < NODE_F) ? (x   + (long)node * NODE_F + c)
                                    : (agg + (long)node * EDGE_F + (c - NODE_F));
    float4 v = *(const float4*)src;
    bf16x4 o;
    o[0] = (__bf16)v.x; o[1] = (__bf16)v.y; o[2] = (__bf16)v.z; o[3] = (__bf16)v.w;
    *(bf16x4*)(h0 + (long)node * D_IN + c) = o;
}

// ---------------------------------------------------------------------------
// 4) weight convert + transpose: W (K x N, f32 row-major) -> Wt (N x K, bf16)
// ---------------------------------------------------------------------------
__global__ void transpose_w_kernel(const float* __restrict__ W,
                                   __bf16*      __restrict__ Wt,
                                   int K, int N) {
    long i = (long)blockIdx.x * blockDim.x + threadIdx.x;
    if (i >= (long)K * N) return;
    int n = (int)(i / K);
    int k = (int)(i % K);
    Wt[i] = (__bf16)W[(long)k * N + n];
}

// ---------------------------------------------------------------------------
// 5) WMMA GEMM:  C[M x N] = act(A[M x K] * Bt[N x K]^T + bias)
//    A, Bt bf16 row-major; acc fp32; C bf16 or fp32.
//    block = 256 threads = 8 waves (2 x 4); wave tile 64x64; block tile 128x256
//    N must be a multiple of 256; K a multiple of 32; M guarded (store side).
//    - Out-of-range A rows are CLAMPED to M-1 (valid memory, discarded at
//      store) so the K loop has no divergent branches.
//    - All addresses are expressions rooted at the __restrict__ kernel args
//      (A + offset) so LLVM keeps them in the GLOBAL address space
//      (global_load_b128, LOADcnt only) instead of falling back to FLAT.
//    - do/while K loop: K >= 32 always, so no zero-trip fallback path.
//    - NOTE: explicit double-buffering was tried and reverted: it pushed the
//      allocator past 256 VGPRs (s_set_vgpr_msb storm + v_mov_b64 copies).
//      The compiler's own pipelining (staggered s_wait_loadcnt, loads sunk
//      between WMMA groups) with ~200 VGPRs is the better schedule; the
//      4 hazard v_nops per WMMA group are co-exec slots that overlap with
//      matrix-pipe execution.
// ---------------------------------------------------------------------------
__device__ __forceinline__ bf16x16 load_frag_bf16(const __bf16* p) {
    bf16x8 lo = *(const bf16x8*)p;          // K = sel*8 .. sel*8+7
    bf16x8 hi = *(const bf16x8*)(p + 16);   // K = 16+sel*8 .. 16+sel*8+7
    return __builtin_shufflevector(lo, hi, 0, 1, 2, 3, 4, 5, 6, 7,
                                           8, 9, 10, 11, 12, 13, 14, 15);
}

template <bool ACT, bool OUT_BF16>
__launch_bounds__(256)
__global__ void gemm_bias_act_kernel(const __bf16* __restrict__ A,
                                     const __bf16* __restrict__ Bt,
                                     const float*  __restrict__ bias,
                                     void*         __restrict__ Cout,
                                     int M, int N, int K) {
    const int lane = threadIdx.x & 31;
    const int wave = threadIdx.x >> 5;
    const int wm   = wave & 1;        // 0..1
    const int wn   = wave >> 1;       // 0..3
    const int mb   = blockIdx.x * 128 + wm * 64;
    const int nb   = blockIdx.y * 256 + wn * 64;
    const int r    = lane & 15;
    const int hx   = lane >> 4;       // which K-half this lane holds

    // Loop-invariant fragment base offsets (elements), rooted at A / Bt.
    long aBase[4];
    long bBase[4];
    #pragma unroll
    for (int i = 0; i < 4; ++i) {
        int row = mb + i * 16 + r;
        row = (row < M) ? row : (M - 1);          // clamp, discard at store
        aBase[i] = (long)row * K + hx * 8;
    }
    #pragma unroll
    for (int j = 0; j < 4; ++j) {
        int col = nb + j * 16 + r;                // N multiple of 256: in range
        bBase[j] = (long)col * K + hx * 8;
    }

    floatx8 acc[4][4] = {};

    int kb = 0;
    do {
        bf16x16 am[4], bn[4];
        #pragma unroll
        for (int i = 0; i < 4; ++i) am[i] = load_frag_bf16(A + aBase[i] + kb);
        #pragma unroll
        for (int j = 0; j < 4; ++j) bn[j] = load_frag_bf16(Bt + bBase[j] + kb);
        #pragma unroll
        for (int i = 0; i < 4; ++i) {
            #pragma unroll
            for (int j = 0; j < 4; ++j) {
                acc[i][j] = __builtin_amdgcn_wmma_f32_16x16x32_bf16(
                    /*neg_a=*/false, am[i], /*neg_b=*/false, bn[j],
                    /*c_mod=*/(short)0, acc[i][j],
                    /*reuse_a=*/false, /*reuse_b=*/false);
            }
        }
        kb += 32;
    } while (kb < K);

    // epilogue: bias + LeakyReLU(0.01), store (lane-coalesced across columns)
    #pragma unroll
    for (int j = 0; j < 4; ++j) {
        int col = nb + j * 16 + r;
        float bv = bias[col];
        #pragma unroll
        for (int i = 0; i < 4; ++i) {
            int rowBase = mb + i * 16 + hx * 8;
            #pragma unroll
            for (int e = 0; e < 8; ++e) {
                int row = rowBase + e;
                if (row < M) {
                    float v = acc[i][j][e] + bv;
                    if (ACT) v = (v > 0.0f) ? v : 0.01f * v;
                    if (OUT_BF16)
                        ((__bf16*)Cout)[(long)row * N + col] = (__bf16)v;
                    else
                        ((float*)Cout)[(long)row * N + col] = v;
                }
            }
        }
    }
}

// ---------------------------------------------------------------------------
// Launch
// ---------------------------------------------------------------------------
extern "C" void kernel_launch(void* const* d_in, const int* in_sizes, int n_in,
                              void* d_out, int out_size, void* d_ws, size_t ws_size,
                              hipStream_t stream) {
    (void)in_sizes; (void)n_in; (void)out_size; (void)ws_size;

    const float* x          = (const float*)d_in[0];
    const int*   edge_index = (const int*)  d_in[1];   // (2, N_EDGES); row = [0..N_EDGES)
    const float* edge_attr  = (const float*)d_in[2];
    // d_in[3] = u, d_in[4] = batch : unused by node MLP
    const float* W1 = (const float*)d_in[5];
    const float* b1 = (const float*)d_in[6];
    const float* W2 = (const float*)d_in[7];
    const float* b2 = (const float*)d_in[8];
    const float* W3 = (const float*)d_in[9];
    const float* b3 = (const float*)d_in[10];
    float* out = (float*)d_out;

    // workspace carve-up (all sizes multiples of 16 B)
    char* ws = (char*)d_ws;
    float*  agg = (float*)ws;  ws += (size_t)N_NODES * EDGE_F * sizeof(float);   // 20.48 MB
    __bf16* h0  = (__bf16*)ws; ws += (size_t)N_NODES * D_IN   * sizeof(__bf16);  // 20.48 MB
    __bf16* h1  = (__bf16*)ws; ws += (size_t)N_NODES * HIDDEN * sizeof(__bf16);  // 81.92 MB
    __bf16* h2  = (__bf16*)ws; ws += (size_t)N_NODES * HIDDEN * sizeof(__bf16);  // 81.92 MB
    __bf16* W1t = (__bf16*)ws; ws += (size_t)D_IN   * HIDDEN * sizeof(__bf16);
    __bf16* W2t = (__bf16*)ws; ws += (size_t)HIDDEN * HIDDEN * sizeof(__bf16);
    __bf16* W3t = (__bf16*)ws; ws += (size_t)HIDDEN * OUT_F  * sizeof(__bf16);

    const int TPB = 256;

    // 1) zero agg
    {
        long n4 = (long)N_NODES * EDGE_F / 4;
        zero_f4_kernel<<<(unsigned)((n4 + TPB - 1) / TPB), TPB, 0, stream>>>((float4*)agg, n4);
    }
    // 2) scatter-add
    {
        long threads = (long)N_EDGES * 32;
        scatter_add_kernel<<<(unsigned)((threads + TPB - 1) / TPB), TPB, 0, stream>>>(
            edge_attr, edge_index, agg, N_EDGES);
    }
    // 3) pack concat -> bf16
    {
        long threads = (long)N_NODES * (D_IN / 4);
        pack_h0_kernel<<<(unsigned)((threads + TPB - 1) / TPB), TPB, 0, stream>>>(
            x, agg, h0, N_NODES);
    }
    // 4) weight transpose + bf16 convert
    {
        long n;
        n = (long)D_IN * HIDDEN;
        transpose_w_kernel<<<(unsigned)((n + TPB - 1) / TPB), TPB, 0, stream>>>(W1, W1t, D_IN, HIDDEN);
        n = (long)HIDDEN * HIDDEN;
        transpose_w_kernel<<<(unsigned)((n + TPB - 1) / TPB), TPB, 0, stream>>>(W2, W2t, HIDDEN, HIDDEN);
        n = (long)HIDDEN * OUT_F;
        transpose_w_kernel<<<(unsigned)((n + TPB - 1) / TPB), TPB, 0, stream>>>(W3, W3t, HIDDEN, OUT_F);
    }
    // 5) three WMMA GEMMs
    {
        dim3 blk(256);
        dim3 g1((N_NODES + 127) / 128, HIDDEN / 256);
        gemm_bias_act_kernel<true,  true ><<<g1, blk, 0, stream>>>(h0, W1t, b1, h1, N_NODES, HIDDEN, D_IN);
        dim3 g2((N_NODES + 127) / 128, HIDDEN / 256);
        gemm_bias_act_kernel<true,  true ><<<g2, blk, 0, stream>>>(h1, W2t, b2, h2, N_NODES, HIDDEN, HIDDEN);
        dim3 g3((N_NODES + 127) / 128, OUT_F / 256);
        gemm_bias_act_kernel<false, false><<<g3, blk, 0, stream>>>(h2, W3t, b3, out, N_NODES, OUT_F, HIDDEN);
    }
}
